// MIL_GCN_Attention_61753039781954
// MI455X (gfx1250) — compile-verified, matrix operand order checked by reading, not run on hardware
//
#include <hip/hip_runtime.h>
#include <hip/hip_bf16.h>

// ---------------------------------------------------------------------------
// MIL-GCN-Attention for MI455X (gfx1250, wave32, WMMA)
// GEMMs use v_wmma_f32_16x16x32_bf16 (f32 inputs converted to bf16 in LDS,
// f32 accumulate). Edge aggregation via global_atomic_add_f32 scatter.
// ---------------------------------------------------------------------------

typedef __attribute__((ext_vector_type(16))) __bf16 v16bf;
typedef __attribute__((ext_vector_type(8)))  __bf16 v8bf;
typedef __attribute__((ext_vector_type(8)))  float  v8f;
typedef __attribute__((ext_vector_type(4)))  float  v4f;

#define TM 128
#define TN 128
#define TK 32
#define LDS_STRIDE 40   // 32 + 8 bf16 pad -> 80B row stride, kills bank conflicts

// ------------------------------ GEMM (WMMA) --------------------------------
// C[M,Nout] = A[M,K] @ W[K,Nout] (+bias) (optional leaky-relu 0.01)
// Requires K % 32 == 0, Nout % 128 == 0. M handled with guards.
__global__ __launch_bounds__(256)
void gemm_bf16_wmma(const float* __restrict__ A, const float* __restrict__ W,
                    const float* __restrict__ bias, float* __restrict__ C,
                    int M, int K, int Nout, int apply_leaky)
{
    __shared__ __bf16 lA[TM * LDS_STRIDE];          // [row][k]
    __shared__ __bf16 lB[TN * LDS_STRIDE];          // transposed: [col][k]

    const int tid  = threadIdx.x;
    const int lane = tid & 31;
    const int wid  = tid >> 5;                       // 0..7 (8 waves)
    const int bm   = blockIdx.y * TM;
    const int bn   = blockIdx.x * TN;

    const int wm = (wid >> 1) * 32;                  // wave row offset: 0/32/64/96
    const int wn = (wid & 1) * 64;                   // wave col offset: 0/64

    v8f acc[2][4];
#pragma unroll
    for (int i = 0; i < 2; ++i)
#pragma unroll
        for (int j = 0; j < 4; ++j)
#pragma unroll
            for (int e = 0; e < 8; ++e) acc[i][j][e] = 0.0f;

    const int lr  = lane & 15;
    const int kbl = (lane < 16) ? 0 : 8;             // per-ISA 16-bit operand layout

    for (int kb = 0; kb < K; kb += TK) {
        // ---- stage A tile: 128x32 f32 -> bf16 (1024 float4, 4 per thread) ----
#pragma unroll
        for (int i = 0; i < 4; ++i) {
            int idx4 = tid + i * 256;                // 0..1023
            int row  = idx4 >> 3;                    // 8 float4 per row
            int kk   = (idx4 & 7) << 2;
            int grow = bm + row;
            v4f v;
            if (grow < M) {
                v = *(const v4f*)(A + (size_t)grow * K + kb + kk);
            } else {
                v[0] = v[1] = v[2] = v[3] = 0.0f;
            }
            __bf16* p = &lA[row * LDS_STRIDE + kk];
            p[0] = (__bf16)v[0]; p[1] = (__bf16)v[1];
            p[2] = (__bf16)v[2]; p[3] = (__bf16)v[3];
        }
        // ---- stage W tile: 32x128 f32 -> bf16, transposed into lB[col][k] ----
#pragma unroll
        for (int i = 0; i < 4; ++i) {
            int idx4 = tid + i * 256;
            int kk   = idx4 >> 5;                    // 32 float4 per k-row
            int c4   = (idx4 & 31) << 2;
            v4f v = *(const v4f*)(W + (size_t)(kb + kk) * Nout + bn + c4);
            lB[(c4 + 0) * LDS_STRIDE + kk] = (__bf16)v[0];
            lB[(c4 + 1) * LDS_STRIDE + kk] = (__bf16)v[1];
            lB[(c4 + 2) * LDS_STRIDE + kk] = (__bf16)v[2];
            lB[(c4 + 3) * LDS_STRIDE + kk] = (__bf16)v[3];
        }
        __syncthreads();

        // ---- load fragments (two ds_load_b128 each) ----
        v16bf afrag[2], bfrag[4];
#pragma unroll
        for (int mt = 0; mt < 2; ++mt) {
            const __bf16* base = &lA[(wm + mt * 16 + lr) * LDS_STRIDE + kbl];
            v8bf lo = *(const v8bf*)(base);
            v8bf hi = *(const v8bf*)(base + 16);
#pragma unroll
            for (int j = 0; j < 8; ++j) { afrag[mt][j] = lo[j]; afrag[mt][8 + j] = hi[j]; }
        }
#pragma unroll
        for (int nt = 0; nt < 4; ++nt) {
            const __bf16* base = &lB[(wn + nt * 16 + lr) * LDS_STRIDE + kbl];
            v8bf lo = *(const v8bf*)(base);
            v8bf hi = *(const v8bf*)(base + 16);
#pragma unroll
            for (int j = 0; j < 8; ++j) { bfrag[nt][j] = lo[j]; bfrag[nt][8 + j] = hi[j]; }
        }

        // ---- 8 WMMAs per wave per K step ----
#pragma unroll
        for (int mt = 0; mt < 2; ++mt)
#pragma unroll
            for (int nt = 0; nt < 4; ++nt)
                acc[mt][nt] = __builtin_amdgcn_wmma_f32_16x16x32_bf16(
                    false, afrag[mt], false, bfrag[nt],
                    (short)0, acc[mt][nt], false, false);

        __syncthreads();
    }

    // ---- epilogue: C/D layout -> lane holds col (lane%16), VGPR r = row r(+8) ----
    const int rsel = (lane < 16) ? 0 : 8;
#pragma unroll
    for (int mt = 0; mt < 2; ++mt)
#pragma unroll
        for (int nt = 0; nt < 4; ++nt) {
            int col = bn + wn + nt * 16 + lr;
            float bv = bias ? bias[col] : 0.0f;
#pragma unroll
            for (int r = 0; r < 8; ++r) {
                int row = bm + wm + mt * 16 + rsel + r;
                if (row < M) {
                    float v = acc[mt][nt][r] + bv;
                    if (apply_leaky) v = v > 0.0f ? v : 0.01f * v;
                    C[(size_t)row * Nout + col] = v;
                }
            }
        }
}

// ------------------------------ small kernels ------------------------------

__global__ void fill_zero(float* __restrict__ p, int n) {
    int i = blockIdx.x * 256 + threadIdx.x;
    if (i < n) p[i] = 0.0f;
}

__global__ void count_deg(const int* __restrict__ dst, int E, float* __restrict__ deg) {
    int e = blockIdx.x * 256 + threadIdx.x;
    if (e < E) atomicAdd(&deg[dst[e]], 1.0f);
}

__global__ void calc_dinv(const float* __restrict__ deg, float* __restrict__ dinv, int N) {
    int i = blockIdx.x * 256 + threadIdx.x;
    if (i < N) dinv[i] = rsqrtf(deg[i] + 1.0f);
}

// agg = h * dinv(i)^2 + bias   (self-loop term + bias, full init of agg)
__global__ void self_init(const float* __restrict__ h, const float* __restrict__ dinv,
                          const float* __restrict__ b, float* __restrict__ agg, int N) {
    size_t idx = (size_t)blockIdx.x * 256 + threadIdx.x;
    int i = (int)(idx >> 9);
    int d = (int)(idx & 511);
    if (i < N) {
        float dv = dinv[i];
        agg[idx] = h[idx] * dv * dv + b[d];
    }
}

// one wave per edge: agg[dst] += h[src] * dinv[src]*dinv[dst]
__global__ __launch_bounds__(256)
void scatter_edges(const int* __restrict__ src, const int* __restrict__ dst,
                   const float* __restrict__ dinv, const float* __restrict__ h,
                   float* __restrict__ agg, int E) {
    int wid  = threadIdx.x >> 5;
    int lane = threadIdx.x & 31;
    int e = blockIdx.x * 8 + wid;
    if (e >= E) return;
    int s = src[e], d = dst[e];
    float coef = dinv[s] * dinv[d];
    const float* hr = h + (size_t)s * 512;
    float* ar = agg + (size_t)d * 512;
#pragma unroll
    for (int j = 0; j < 16; ++j) {
        int c = lane + j * 32;
        atomicAdd(&ar[c], hr[c] * coef);
    }
}

__global__ void leaky_ip(float* __restrict__ x, size_t n) {
    size_t i = (size_t)blockIdx.x * 256 + threadIdx.x;
    if (i < n) {
        float v = x[i];
        x[i] = v > 0.0f ? v : 0.01f * v;
    }
}

// logits[i] = dot(s1[i,0:128], Wa2) + ba2 ; one wave per node
__global__ __launch_bounds__(256)
void att_logits(const float* __restrict__ s1, const float* __restrict__ Wa2,
                const float* __restrict__ ba2, float* __restrict__ logits, int N) {
    int wid  = threadIdx.x >> 5;
    int lane = threadIdx.x & 31;
    int i = blockIdx.x * 8 + wid;
    if (i >= N) return;
    const float* r = s1 + (size_t)i * 128;
    float s = 0.0f;
#pragma unroll
    for (int j = 0; j < 4; ++j) s += r[lane + j * 32] * Wa2[lane + j * 32];
#pragma unroll
    for (int off = 16; off > 0; off >>= 1) s += __shfl_down(s, off, 32);
    if (lane == 0) logits[i] = s + ba2[0];
}

__global__ void reduce_max_part(const float* __restrict__ x, int n, float* __restrict__ part) {
    __shared__ float sm[256];
    float m = -3.4e38f;
    for (int i = blockIdx.x * 256 + threadIdx.x; i < n; i += gridDim.x * 256)
        m = fmaxf(m, x[i]);
    sm[threadIdx.x] = m; __syncthreads();
    for (int s = 128; s > 0; s >>= 1) {
        if (threadIdx.x < s) sm[threadIdx.x] = fmaxf(sm[threadIdx.x], sm[threadIdx.x + s]);
        __syncthreads();
    }
    if (threadIdx.x == 0) part[blockIdx.x] = sm[0];
}

__global__ void final_max(const float* __restrict__ part, int n, float* __restrict__ out) {
    __shared__ float sm[256];
    float m = -3.4e38f;
    for (int i = threadIdx.x; i < n; i += 256) m = fmaxf(m, part[i]);
    sm[threadIdx.x] = m; __syncthreads();
    for (int s = 128; s > 0; s >>= 1) {
        if (threadIdx.x < s) sm[threadIdx.x] = fmaxf(sm[threadIdx.x], sm[threadIdx.x + s]);
        __syncthreads();
    }
    if (threadIdx.x == 0) out[0] = sm[0];
}

__global__ void exp_partial(const float* __restrict__ logits, const float* __restrict__ gmax,
                            float* __restrict__ a_out, int n, float* __restrict__ part) {
    __shared__ float sm[256];
    float mx = *gmax;
    float s = 0.0f;
    for (int i = blockIdx.x * 256 + threadIdx.x; i < n; i += gridDim.x * 256) {
        float e = expf(logits[i] - mx);
        a_out[i] = e;
        s += e;
    }
    sm[threadIdx.x] = s; __syncthreads();
    for (int st = 128; st > 0; st >>= 1) {
        if (threadIdx.x < st) sm[threadIdx.x] += sm[threadIdx.x + st];
        __syncthreads();
    }
    if (threadIdx.x == 0) part[blockIdx.x] = sm[0];
}

__global__ void final_sum(const float* __restrict__ part, int n, float* __restrict__ out) {
    __shared__ float sm[256];
    float s = 0.0f;
    for (int i = threadIdx.x; i < n; i += 256) s += part[i];
    sm[threadIdx.x] = s; __syncthreads();
    for (int st = 128; st > 0; st >>= 1) {
        if (threadIdx.x < st) sm[threadIdx.x] += sm[threadIdx.x + st];
        __syncthreads();
    }
    if (threadIdx.x == 0) out[0] = sm[0];
}

__global__ void normalize_a(float* __restrict__ a, const float* __restrict__ gsum, int n) {
    int i = blockIdx.x * 256 + threadIdx.x;
    if (i < n) a[i] = a[i] / (*gsum);
}

// z[d] += sum over this block's row chunk of a[r]*h[r,d]; thread t owns cols t, t+256
__global__ __launch_bounds__(256)
void pool_attn(const float* __restrict__ a, const float* __restrict__ h,
               float* __restrict__ z, int N) {
    int t = threadIdx.x;
    float acc0 = 0.0f, acc1 = 0.0f;
    int r0 = blockIdx.x * 256;
    int rend = r0 + 256; if (rend > N) rend = N;
    for (int r = r0; r < rend; ++r) {
        float av = a[r];
        acc0 += av * h[(size_t)r * 512 + t];
        acc1 += av * h[(size_t)r * 512 + t + 256];
    }
    atomicAdd(&z[t], acc0);
    atomicAdd(&z[t + 256], acc1);
}

__global__ void final_out(const float* __restrict__ z, const float* __restrict__ Wc,
                          const float* __restrict__ bc, float* __restrict__ out) {
    __shared__ float sm[256];
    int t = threadIdx.x;
    sm[t] = z[t] * Wc[t] + z[t + 256] * Wc[t + 256];
    __syncthreads();
    for (int s = 128; s > 0; s >>= 1) {
        if (t < s) sm[t] += sm[t + s];
        __syncthreads();
    }
    if (t == 0) out[0] = sm[0] + bc[0];
}

// ------------------------------- launcher ----------------------------------

extern "C" void kernel_launch(void* const* d_in, const int* in_sizes, int n_in,
                              void* d_out, int out_size, void* d_ws, size_t ws_size,
                              hipStream_t stream) {
    const float* x   = (const float*)d_in[0];
    const int*   ei  = (const int*)d_in[1];
    const float* W1  = (const float*)d_in[2];
    const float* b1  = (const float*)d_in[3];
    const float* W2  = (const float*)d_in[4];
    const float* b2  = (const float*)d_in[5];
    const float* Wa1 = (const float*)d_in[6];
    const float* ba1 = (const float*)d_in[7];
    const float* Wa2 = (const float*)d_in[8];
    const float* ba2 = (const float*)d_in[9];
    const float* Wc  = (const float*)d_in[10];
    const float* bc  = (const float*)d_in[11];

    const int N = in_sizes[0] / 512;        // 50000
    const int E = in_sizes[1] / 2;          // 400000
    const int D = 512;
    const size_t MP = (size_t)((N + 127) / 128) * 128;   // padded rows (50048)

    const int* src = ei;
    const int* dst = ei + E;

    float* out   = (float*)d_out;           // [0]
    float* a_out = out + 1;                 // [1 .. N]
    float* hout  = a_out + N;               // [1+N ..) : N x 512, layer-2 output

    float* bufA   = (float*)d_ws;           // MP*512  (GEMM outputs / s1)
    float* bufB   = bufA + MP * D;          // MP*512  (layer-1 hidden)
    float* deg    = bufB + MP * D;          // MP
    float* dinv   = deg + MP;               // MP
    float* logits = dinv + MP;              // MP
    float* part   = logits + MP;            // 1024
    float* scal   = part + 1024;            // gmax, gsum
    float* zbuf   = scal + 8;               // 512

    const size_t NH = (size_t)N * D;
    const int gElem = (int)((NH + 255) / 256);        // elementwise over N*512
    const int gN    = (N + 255) / 256;
    const int gE    = (E + 255) / 256;
    const int gEdgeWave = (E + 7) / 8;
    const dim3 gGemm(D / 128, (unsigned)(MP / 128));  // 4 x 391
    const dim3 gGemmAtt(1, (unsigned)(MP / 128));     // 1 x 391

    // ---- degree / normalization ----
    fill_zero<<<(int)((MP + 255) / 256), 256, 0, stream>>>(deg, (int)MP);
    fill_zero<<<2, 256, 0, stream>>>(zbuf, 512);
    count_deg<<<gE, 256, 0, stream>>>(dst, E, deg);
    calc_dinv<<<gN, 256, 0, stream>>>(deg, dinv, N);

    // ---- layer 1: h = leaky(GCNConv(x, W1, b1)) ----
    gemm_bf16_wmma<<<gGemm, 256, 0, stream>>>(x, W1, nullptr, bufA, N, D, D, 0);
    self_init<<<gElem, 256, 0, stream>>>(bufA, dinv, b1, bufB, N);
    scatter_edges<<<gEdgeWave, 256, 0, stream>>>(src, dst, dinv, bufA, bufB, E);
    leaky_ip<<<gElem, 256, 0, stream>>>(bufB, NH);

    // ---- layer 2: h = leaky(GCNConv(h, W2, b2)) -> directly into d_out ----
    gemm_bf16_wmma<<<gGemm, 256, 0, stream>>>(bufB, W2, nullptr, bufA, N, D, D, 0);
    self_init<<<gElem, 256, 0, stream>>>(bufA, dinv, b2, hout, N);
    scatter_edges<<<gEdgeWave, 256, 0, stream>>>(src, dst, dinv, bufA, hout, E);
    leaky_ip<<<gElem, 256, 0, stream>>>(hout, NH);

    // ---- attention MLP: s1 = leaky(h@Wa1+ba1), logit = s1@Wa2+ba2 ----
    gemm_bf16_wmma<<<gGemmAtt, 256, 0, stream>>>(hout, Wa1, ba1, bufA, N, D, 128, 1);
    att_logits<<<(N + 7) / 8, 256, 0, stream>>>(bufA, Wa2, ba2, logits, N);

    // ---- softmax over nodes ----
    reduce_max_part<<<256, 256, 0, stream>>>(logits, N, part);
    final_max<<<1, 256, 0, stream>>>(part, 256, &scal[0]);
    exp_partial<<<256, 256, 0, stream>>>(logits, &scal[0], a_out, N, part);
    final_sum<<<1, 256, 0, stream>>>(part, 256, &scal[1]);
    normalize_a<<<gN, 256, 0, stream>>>(a_out, &scal[1], N);

    // ---- attention pooling + final projection ----
    pool_attn<<<(N + 255) / 256, 256, 0, stream>>>(a_out, hout, zbuf, N);
    final_out<<<1, 256, 0, stream>>>(zbuf, Wc, bc, out);
}
